// NeuralTensorLayer_4045859193029
// MI455X (gfx1250) — matrix operand order repaired and checked by reading, not compiled, not run on hardware
//
#include <hip/hip_runtime.h>
#include <hip/hip_bf16.h>

#define BATCH 65536
#define DDIM  128
#define KOUT  64

typedef __attribute__((ext_vector_type(16))) _Float16 v16h;
typedef __attribute__((ext_vector_type(8)))  _Float16 v8h;
typedef __attribute__((ext_vector_type(8)))  float    v8f;

// ---- workspace layout (bytes) ----
// Wfrag: KOUT*DDIM*DDIM f16, fragment-ordered          = 2,097,152
// Vfrag: 2*DDIM*KOUT   f16, fragment-ordered           =    32,768
// sumb : 1 float
#define WFRAG_OFF 0
#define VFRAG_OFF 2097152
#define SUMB_OFF  (2097152 + 32768)

// ---------------------------------------------------------------------------
// Fragment-layout conventions (this code's k order, wave32):
//  A (16x32 f16): lane l -> M = l&15, half h = l>>4;
//     halfs 0..7  hold K = kbase + 16h + q
//     halfs 8..15 hold K = kbase + 16h + 8 + q   (contiguous per lane)
//  B (32x16 f16): lane l -> N = l&15, half h = l>>4;
//     half  hh    holds K = kbase + 16h + hh    (contiguous per lane)
//  C/D (16x16 f32): lane l -> N = l&15; VGPR r -> M = r + 8h
// ---------------------------------------------------------------------------

__device__ __forceinline__ void wait_asynccnt0() {
#if __has_builtin(__builtin_amdgcn_s_wait_asynccnt)
  __builtin_amdgcn_s_wait_asynccnt(0);
#else
  asm volatile("s_wait_asynccnt 0x0" ::: "memory");
#endif
}

// Async-stage one 32 KB W slice (2048 x 16B) into LDS via the CDNA5
// GLOBAL_LOAD_ASYNC_TO_LDS path (ASYNCcnt, no data VGPRs). Per ISA §15.18
// the instruction offset applies to both the LDS and global addresses, but we
// fold the per-iteration stride into the addresses to keep the asm trivial.
__device__ __forceinline__ void stage_async(const _Float16* __restrict__ src,
                                            uint32_t lds_off, int tid) {
  uint64_t g = (uint64_t)(uintptr_t)src + (uint64_t)tid * 16u;
  uint32_t l = lds_off + (uint32_t)tid * 16u;
#pragma unroll
  for (int i = 0; i < 8; ++i) {
    asm volatile("global_load_async_to_lds_b128 %0, %1, off"
                 :: "v"(l + (uint32_t)(i * 4096)),
                    "v"(g + (uint64_t)(i * 4096))
                 : "memory");
  }
}

// Pre-convert W and V (f32, row-major) into fragment-ordered f16 so the hot
// loop's B-fragment loads are contiguous 32B per lane. Also sum the bias.
__global__ __launch_bounds__(256) void ntl_convert(
    const float* __restrict__ W, const float* __restrict__ V,
    const float* __restrict__ b, _Float16* __restrict__ Wfrag,
    _Float16* __restrict__ Vfrag, float* __restrict__ sumb) {
  const int t = blockIdx.x * 256 + threadIdx.x;
  const int NW = KOUT * DDIM * DDIM;          // 1,048,576
  const int NV = 2 * DDIM * KOUT;             // 16,384
  if (t < NW) {
    // layout: [k][jt(8)][kt(4)][lane(32)][hh(16)]
    const int hh = t & 15, lane = (t >> 4) & 31;
    const int kt = (t >> 9) & 3, jt = (t >> 11) & 7, k = t >> 14;
    const int i = kt * 32 + ((lane >> 4) << 4) + hh;   // K index into W rows
    const int j = jt * 16 + (lane & 15);               // N index into W cols
    Wfrag[t] = (_Float16)W[((size_t)k * DDIM + i) * DDIM + j];
  } else if (t < NW + NV) {
    // layout: [kt(8)][nt(4)][lane(32)][hh(16)]
    const int u = t - NW;
    const int hh = u & 15, lane = (u >> 4) & 31;
    const int nt = (u >> 9) & 3, kt = u >> 11;
    const int i = kt * 32 + ((lane >> 4) << 4) + hh;   // row of V (0..255)
    const int j = nt * 16 + (lane & 15);               // col of V (0..63)
    Vfrag[u] = (_Float16)V[(size_t)i * KOUT + j];
  } else if (t == NW + NV) {
    float s = 0.f;
    for (int i = 0; i < DDIM; ++i) s += b[i];
    sumb[0] = s;
  }
}

// Load one A fragment (16x32 f16) for this lane straight from f32 global,
// converting in registers. rowptr = &mat[row*DDIM].
__device__ __forceinline__ v16h load_a_frag_f32(const float* __restrict__ rowptr,
                                                int kbase, int h) {
  const float4* p1 = (const float4*)(rowptr + kbase + 16 * h);
  const float4* p2 = (const float4*)(rowptr + kbase + 16 * h + 8);
  float4 a0 = p1[0], a1 = p1[1], b0 = p2[0], b1 = p2[1];
  v16h r;
  r[0]  = (_Float16)a0.x; r[1]  = (_Float16)a0.y; r[2]  = (_Float16)a0.z; r[3]  = (_Float16)a0.w;
  r[4]  = (_Float16)a1.x; r[5]  = (_Float16)a1.y; r[6]  = (_Float16)a1.z; r[7]  = (_Float16)a1.w;
  r[8]  = (_Float16)b0.x; r[9]  = (_Float16)b0.y; r[10] = (_Float16)b0.z; r[11] = (_Float16)b0.w;
  r[12] = (_Float16)b1.x; r[13] = (_Float16)b1.y; r[14] = (_Float16)b1.z; r[15] = (_Float16)b1.w;
  return r;
}

__device__ __forceinline__ v16h load_b_frag_lds(const _Float16* buf, int frag, int lane) {
  const v8h* p = (const v8h*)(buf + ((size_t)(frag * 32 + lane)) * 16);
  v8h lo = p[0], hi = p[1];
  v16h r;
#pragma unroll
  for (int q = 0; q < 8; ++q) { r[q] = lo[q]; r[q + 8] = hi[q]; }
  return r;
}

// ---------------------------------------------------------------------------
// Bilinear term: btp[k][b] = sum_ij e1[b,i] W[k,i,j] e2[b,j], stored flat
// [k][B] directly into d_out (the reference's reshape is a flat reinterpret).
// Grid: 512 WGs x 256 thr. WG owns 128 batch rows; wave w owns rows 16w..16w+15.
// W slices double-buffered in LDS via async-to-LDS, overlapped with WMMA.
// ---------------------------------------------------------------------------
__global__ __launch_bounds__(256) void ntl_bilinear(
    const float* __restrict__ e1, const float* __restrict__ e2,
    const _Float16* __restrict__ Wfrag, float* __restrict__ btp) {
  __shared__ __align__(16) _Float16 wbuf[2][16384];  // 2 x 32 KB, double buffer

  const int tid  = threadIdx.x;
  const int lane = tid & 31;
  const int w    = tid >> 5;
  const int h    = lane >> 4;
  const int nl   = lane & 15;
  const int bb   = blockIdx.x * 128 + 16 * w;  // wave's batch base
  const int arow = bb + nl;                    // this lane's A-matrix row

  // LDS byte offsets of the two W buffers (low 32 bits of a generic LDS ptr).
  const uint32_t wb0 = (uint32_t)(uintptr_t)&wbuf[0][0];
  const uint32_t wb1 = (uint32_t)(uintptr_t)&wbuf[1][0];

  // e1 A-fragments, K = 0..127 in 4 chunks; resident for the whole k-loop.
  v16h afrag[4];
#pragma unroll
  for (int kt = 0; kt < 4; ++kt)
    afrag[kt] = load_a_frag_f32(e1 + (size_t)arow * DDIM, kt * 32, h);

  // e2 in C-register layout: e2c[jt][r] = e2[bb + r + 8h, jt*16 + nl]
  float e2c[8][8];
#pragma unroll
  for (int jt = 0; jt < 8; ++jt)
#pragma unroll
    for (int r = 0; r < 8; ++r)
      e2c[jt][r] = e2[(size_t)(bb + r + 8 * h) * DDIM + jt * 16 + nl];

  // prologue: async-stage W[0]
  stage_async(Wfrag, wb0, tid);
  wait_asynccnt0();
  __syncthreads();

  for (int k = 0; k < KOUT; ++k) {
    const _Float16* cur = &wbuf[k & 1][0];
    if (k + 1 < KOUT)  // fire-and-forget stage of the next slice
      stage_async(Wfrag + (size_t)(k + 1) * 16384, (k & 1) ? wb0 : wb1, tid);

    float racc[8] = {0.f, 0.f, 0.f, 0.f, 0.f, 0.f, 0.f, 0.f};
#pragma unroll
    for (int jt = 0; jt < 8; ++jt) {
      v8f c = {0.f, 0.f, 0.f, 0.f, 0.f, 0.f, 0.f, 0.f};
#pragma unroll
      for (int kt = 0; kt < 4; ++kt) {
        v16h bfrag = load_b_frag_lds(cur, jt * 4 + kt, lane);
        c = __builtin_amdgcn_wmma_f32_16x16x32_f16(
            false, afrag[kt], false, bfrag, (short)0, c, false, false);
      }
      // row-wise dot with e2 in matching C layout
#pragma unroll
      for (int r = 0; r < 8; ++r) racc[r] += c[r] * e2c[jt][r];
    }

    // reduce partial row sums across the 16 lanes of each half, then store.
#pragma unroll
    for (int r = 0; r < 8; ++r) {
      float s = racc[r];
#pragma unroll
      for (int off = 1; off < 16; off <<= 1) s += __shfl_xor(s, off, 16);
      if (nl == r)  // lane r (h=0) owns row r, lane 16+r (h=1) owns row r+8
        btp[(size_t)k * BATCH + bb + r + 8 * h] = s;
    }

    wait_asynccnt0();   // next slice fully landed in LDS
    __syncthreads();
  }
}

// ---------------------------------------------------------------------------
// feed_forward = [e1|e2] @ V via WMMA (K=256), fused with the flat-reshape
// read of btp (already in d_out) and tanh epilogue.
// ---------------------------------------------------------------------------
__global__ __launch_bounds__(256) void ntl_ff_combine(
    const float* __restrict__ e1, const float* __restrict__ e2,
    const _Float16* __restrict__ Vfrag, const float* __restrict__ sumb,
    float* __restrict__ out) {
  const int tid  = threadIdx.x;
  const int lane = tid & 31;
  const int w    = tid >> 5;
  const int h    = lane >> 4;
  const int nl   = lane & 15;
  const int bb   = blockIdx.x * 128 + 16 * w;
  const int arow = bb + nl;

  v8f zero = {0.f, 0.f, 0.f, 0.f, 0.f, 0.f, 0.f, 0.f};
  v8f c[4];
#pragma unroll
  for (int nt = 0; nt < 4; ++nt) c[nt] = zero;

#pragma unroll
  for (int kt = 0; kt < 8; ++kt) {
    const float* src = (kt < 4) ? (e1 + (size_t)arow * DDIM)
                                : (e2 + (size_t)arow * DDIM);
    v16h a = load_a_frag_f32(src, (kt & 3) * 32, h);
#pragma unroll
    for (int nt = 0; nt < 4; ++nt) {
      const v8h* bp = (const v8h*)(Vfrag + ((size_t)((kt * 4 + nt) * 32 + lane)) * 16);
      v8h lo = bp[0], hi = bp[1];
      v16h bfrag;
#pragma unroll
      for (int q = 0; q < 8; ++q) { bfrag[q] = lo[q]; bfrag[q + 8] = hi[q]; }
      c[nt] = __builtin_amdgcn_wmma_f32_16x16x32_f16(
          false, a, false, bfrag, (short)0, c[nt], false, false);
    }
  }

  const float sb = sumb[0];
#pragma unroll
  for (int nt = 0; nt < 4; ++nt) {
#pragma unroll
    for (int r = 0; r < 8; ++r) {
      const int m = r + 8 * h;
      const size_t idx = (size_t)(bb + m) * KOUT + nt * 16 + nl;
      const float bt = out[idx];                 // btp flat-reinterpreted
      out[idx] = tanhf(c[nt][r] + bt + sb);
    }
  }
}

extern "C" void kernel_launch(void* const* d_in, const int* in_sizes, int n_in,
                              void* d_out, int out_size, void* d_ws, size_t ws_size,
                              hipStream_t stream) {
  const float* e1 = (const float*)d_in[0];
  const float* e2 = (const float*)d_in[1];
  const float* W  = (const float*)d_in[2];
  const float* V  = (const float*)d_in[3];
  const float* b  = (const float*)d_in[4];
  float* out = (float*)d_out;

  char* ws = (char*)d_ws;
  _Float16* Wfrag = (_Float16*)(ws + WFRAG_OFF);
  _Float16* Vfrag = (_Float16*)(ws + VFRAG_OFF);
  float*    sumb  = (float*)(ws + SUMB_OFF);

  // 1) pre-convert W,V to fragment-ordered f16 (+ bias sum)
  const int NTOT = KOUT * DDIM * DDIM + 2 * DDIM * KOUT + 1;
  ntl_convert<<<(NTOT + 255) / 256, 256, 0, stream>>>(W, V, b, Wfrag, Vfrag, sumb);

  // 2) bilinear term -> btp stored flat [k][B] directly in d_out
  ntl_bilinear<<<BATCH / 128, 256, 0, stream>>>(e1, e2, Wfrag, out);

  // 3) feed-forward GEMM + flat-reshape read + tanh epilogue, in place
  ntl_ff_combine<<<BATCH / 128, 256, 0, stream>>>(e1, e2, Vfrag, sumb, out);
}
